// LSTMCell_11192684773681
// MI455X (gfx1250) — matrix operand order repaired and checked by reading, not compiled, not run on hardware
//
#include <hip/hip_runtime.h>

typedef __attribute__((ext_vector_type(16))) _Float16 v16h;
typedef __attribute__((ext_vector_type(8)))  _Float16 v8h;
typedef __attribute__((ext_vector_type(8)))  float    v8f;

// Problem sizes (fixed by the reference)
#define BSZ   1024          // batch
#define USZ   2048          // num_units
#define KDIM  4096          // total GEMM reduction = K * (2U/K)
#define NCOL  8192          // total GEMM output cols = N * (4U/N)

// ---------------------------------------------------------------------------
// CDNA5 async copy: global -> LDS, 16 bytes per lane, tracked by ASYNCcnt.
// lds = wave-relative LDS byte address (low 32 bits of a generic __shared__
// pointer), g = per-lane global address.
// ---------------------------------------------------------------------------
__device__ __forceinline__ void async_b128(unsigned lds, const void* g) {
  asm volatile("global_load_async_to_lds_b128 %0, %1, off"
               :: "v"(lds), "v"(g) : "memory");
}

// ---------------------------------------------------------------------------
// Kernel 1: build XH (f16) [1024 x 4096].
// Column dk = k*2048 + d ; d<1024 -> x[:, k*1024+d], else h[:, k*1024+d-1024]
// ---------------------------------------------------------------------------
__global__ __launch_bounds__(256) void pack_A(const float* __restrict__ x,
                                              const float* __restrict__ h,
                                              _Float16* __restrict__ A) {
  int idx = blockIdx.x * 256 + threadIdx.x;          // 0 .. 1024*4096-1
  int row = idx >> 12;
  int dk  = idx & 4095;
  int k = dk >> 11;
  int d = dk & 2047;
  float v = (d < 1024) ? x[(size_t)row * USZ + (k << 10) + d]
                       : h[(size_t)row * USZ + (k << 10) + (d - 1024)];
  A[idx] = (_Float16)v;
}

// ---------------------------------------------------------------------------
// Kernel 2: W [k][n][d][f] (f32) -> Bt[col][dk] (f16), col = n*4096+f,
// dk = k*2048+d. LDS-tiled 64x64 transpose so both global phases coalesce.
// ---------------------------------------------------------------------------
__global__ __launch_bounds__(256) void pack_W(const float* __restrict__ W,
                                              _Float16* __restrict__ Bt) {
  __shared__ _Float16 tile[64][72];                  // pad to dodge bank conflicts
  const int f0 = blockIdx.x * 64;
  const int d0 = blockIdx.y * 64;
  const int kn = blockIdx.z;                         // k*2 + n
  const int k = kn >> 1, n = kn & 1;
  const int t = threadIdx.x;
#pragma unroll
  for (int j = 0; j < 16; ++j) {
    int idx = t + j * 256;
    int r = idx >> 6, c = idx & 63;
    tile[r][c] = (_Float16)W[((size_t)(kn * 2048 + d0 + r)) * 4096 + f0 + c];
  }
  __syncthreads();
#pragma unroll
  for (int j = 0; j < 16; ++j) {
    int idx = t + j * 256;
    int fr = idx >> 6, dc = idx & 63;
    Bt[(size_t)(n * 4096 + f0 + fr) * KDIM + (size_t)(k * 2048 + d0 + dc)] =
        tile[dc][fr];
  }
}

// ---------------------------------------------------------------------------
// Kernel 3: fused GEMM (f16 WMMA, f32 acc) + LSTM gates + cell update.
// Block = 256 threads = 8 wave32s. Output tile: 64 rows x (4 gates x 64 u).
// Wave (wm,wu): rows [wm*16,+16), u-cols [wu*32,+32) across ALL 4 gates,
// so the gate-fusion epilogue needs no cross-wave traffic.
// Double-buffered LDS staging via global_load_async_to_lds_b128 (ASYNCcnt).
// ---------------------------------------------------------------------------
__global__ __launch_bounds__(256) void lstm_gemm(const _Float16* __restrict__ A,
                                                 const _Float16* __restrict__ Bt,
                                                 const float* __restrict__ cin,
                                                 float* __restrict__ out) {
  __shared__ __align__(16) _Float16 As[2][64][40];   // 64 rows x 32 K (+8 pad)
  __shared__ __align__(16) _Float16 Bs[2][256][40];  // 256 cols x 32 K (+8 pad)

  const int row0 = blockIdx.x * 64;
  const int u0   = blockIdx.y * 64;
  const int n    = blockIdx.z;
  const int t    = threadIdx.x;
  const int lane = t & 31;                           // wave32
  const int wave = t >> 5;                           // 0..7
  const int wm   = wave & 3;                         // row sub-tile (16 rows)
  const int wu   = wave >> 2;                        // u half (32 cols)

  v8f acc[4][2];
#pragma unroll
  for (int g = 0; g < 4; ++g)
#pragma unroll
    for (int tt = 0; tt < 2; ++tt)
#pragma unroll
      for (int i = 0; i < 8; ++i) acc[g][tt][i] = 0.0f;

  // Staging assignments (per thread)
  const int ct = t;                                  // B staging col 0..255
  const size_t bcol = (size_t)(n * 4096 + (ct >> 6) * 1024 + u0 + (ct & 63));
  const int ar = t >> 2;                             // A staging row 0..63
  const int ac = (t & 3) * 8;                        // A staging K offset
  const _Float16* gA = A + (size_t)(row0 + ar) * KDIM + ac;   // + kk
  const _Float16* gB = Bt + bcol * KDIM;                      // + kk + j*8

  // Issue one tile's worth (5 x b128 per thread) of async copies into buf.
  auto prefetch = [&](int buf, int kk) {
    async_b128((unsigned)(uintptr_t)&As[buf][ar][ac], gA + kk);
#pragma unroll
    for (int j = 0; j < 4; ++j)
      async_b128((unsigned)(uintptr_t)&Bs[buf][ct][j * 8], gB + kk + j * 8);
  };

  const int NIT = KDIM / 32;                         // 128 k-steps
  prefetch(0, 0);

  for (int i = 0; i < NIT; ++i) {
    const int cur = i & 1;
    // Prefetch next tile (clamped on the last step; writes the idle buffer).
    const int kn2 = (i + 1 < NIT) ? (i + 1) * 32 : i * 32;
    prefetch(cur ^ 1, kn2);
    // Drain everything except the 5 prefetch ops just issued.
    asm volatile("s_wait_asynccnt 0x5" ::: "memory");
    __syncthreads();

    // A fragment (16x32 f16): lane<16 holds K 0..7 & 16..23 of row lane,
    // lane>=16 holds K 8..15 & 24..31 of row lane-16.
    const int amr = wm * 16 + (lane & 15);
    const int akc = (lane >> 4) * 8;
    v8h alo = *(const v8h*)&As[cur][amr][akc];
    v8h ahi = *(const v8h*)&As[cur][amr][akc + 16];
    v16h a = __builtin_shufflevector(alo, ahi, 0, 1, 2, 3, 4, 5, 6, 7,
                                     8, 9, 10, 11, 12, 13, 14, 15);

    // Load ALL B fragments first (independent registers), then WMMA burst,
    // so DS latency amortizes across the 8 matrix ops.
    v16h bfr[4][2];
#pragma unroll
    for (int g = 0; g < 4; ++g) {
#pragma unroll
      for (int tt = 0; tt < 2; ++tt) {
        const int bc = g * 64 + wu * 32 + tt * 16 + (lane & 15);
        const int bk = (lane >> 4) * 8;
        v8h blo = *(const v8h*)&Bs[cur][bc][bk];
        v8h bhi = *(const v8h*)&Bs[cur][bc][bk + 16];
        bfr[g][tt] = __builtin_shufflevector(blo, bhi, 0, 1, 2, 3, 4, 5, 6, 7,
                                             8, 9, 10, 11, 12, 13, 14, 15);
      }
    }
#pragma unroll
    for (int g = 0; g < 4; ++g)
#pragma unroll
      for (int tt = 0; tt < 2; ++tt)
        acc[g][tt] = __builtin_amdgcn_wmma_f32_16x16x32_f16(
            false, a, false, bfr[g][tt], (short)0, acc[g][tt], false, false);

    __syncthreads();   // everyone done reading buf[cur] before it is re-filled
  }

  // Epilogue: gates + cell update, all in-register (this wave owns i,f,g,o
  // for its rows x u-cols). C/D layout: VGPR v, lane<16 -> M=v,N=lane;
  // lane>=16 -> M=v+8, N=lane-16.
  float* outH = out;
  float* outC = out + (size_t)BSZ * USZ;
#pragma unroll
  for (int tt = 0; tt < 2; ++tt) {
#pragma unroll
    for (int v = 0; v < 8; ++v) {
      const int m   = (lane >> 4) * 8 + v;
      const int row = row0 + wm * 16 + m;
      const int ul  = wu * 32 + tt * 16 + (lane & 15);
      const size_t gi = (size_t)row * USZ + (size_t)(n * 1024 + u0 + ul);
      float ig = acc[0][tt][v], fg = acc[1][tt][v];
      float gg = acc[2][tt][v], og = acc[3][tt][v];
      ig = 1.0f / (1.0f + __expf(-ig));
      fg = 1.0f / (1.0f + __expf(-fg));
      og = 1.0f / (1.0f + __expf(-og));
      gg = 1.0f - 2.0f / (__expf(2.0f * gg) + 1.0f);   // tanh
      float cp = cin[gi];
      float nc = fg * cp + ig * gg;
      float th = 1.0f - 2.0f / (__expf(2.0f * nc) + 1.0f);
      outH[gi] = og * th;
      outC[gi] = nc;
    }
  }
}

// ---------------------------------------------------------------------------
extern "C" void kernel_launch(void* const* d_in, const int* in_sizes, int n_in,
                              void* d_out, int out_size, void* d_ws, size_t ws_size,
                              hipStream_t stream) {
  (void)in_sizes; (void)n_in; (void)out_size; (void)ws_size;
  const float* x = (const float*)d_in[0];
  const float* h = (const float*)d_in[1];
  const float* c = (const float*)d_in[2];
  const float* W = (const float*)d_in[3];
  float* out = (float*)d_out;

  _Float16* A  = (_Float16*)d_ws;                                    // 8 MB
  _Float16* Bt = (_Float16*)((char*)d_ws + (size_t)8 * 1024 * 1024); // 64 MB

  pack_A<<<(BSZ * KDIM) / 256, 256, 0, stream>>>(x, h, A);
  pack_W<<<dim3(64, 32, 4), 256, 0, stream>>>(W, Bt);
  lstm_gemm<<<dim3(BSZ / 64, 1024 / 64, 2), 256, 0, stream>>>(A, Bt, c, out);
}
